// ChainCRF_65781719106080
// MI455X (gfx1250) — matrix operand order repaired and checked by reading, not compiled, not run on hardware
//
#include <hip/hip_runtime.h>

// ============================================================================
// Sparse chain-CRF loss for B=256, T=1024, C=64 on gfx1250 (MI455X).
//
// Free energy (logZ) via forward recursion as exp-domain GEMMs, in BASE-2:
//   a2_t = log2( exp2(a2_{t-1} - m) @ exp2(U*log2e) ) + m + x_t*log2e
// Computed transposed, D[j][b] = expU^T @ exp2(a2)^T:
//   * A operand = expU^T : constant, precomputed into registers (8 fragments)
//   * B operand rebuilt from the D-layout accumulator each step using only
//     v_permlanex16 half-swaps — no LDS, no DS-counter waits.
//   * raw v_exp_f32 / v_log_f32 (no libm denorm guards: acc >= ~0.5 always)
//   * v_cvt_pk_rtz_f16_f32 packs the f16 B operand two-at-a-time
// 8 x v_wmma_f32_16x16x32_f16 per timestep per wave; 16 waves cover B=256.
// ============================================================================

typedef __attribute__((ext_vector_type(16))) _Float16 v16h;
typedef __attribute__((ext_vector_type(8)))  float    v8f;
typedef __attribute__((ext_vector_type(2)))  __fp16   v2fp16;

#define CRF_B 256
#define CRF_T 1024
#define CRF_C 64
#define CRF_LOG2E 1.4426950408889634f
#define CRF_LN2   0.6931471805599453f

union BFrag { v16h v; unsigned u[8]; };
union PK2   { v2fp16 v; unsigned u; };

// xor-16 half-swap as pure VALU (v_permlanex16_b32, identity lane selects)
static __device__ __forceinline__ float xor16f(float v) {
    return __int_as_float(__builtin_amdgcn_permlanex16(
        __float_as_int(v), __float_as_int(v),
        0x76543210, (int)0xfedcba98, false, false));
}
static __device__ __forceinline__ unsigned xor16u(unsigned v) {
    return (unsigned)__builtin_amdgcn_permlanex16(
        (int)v, (int)v, 0x76543210, (int)0xfedcba98, false, false);
}

// Load one timestep of emissions (pre-scaled by log2e) into registers.
static __device__ __forceinline__
void crf_load_x(float (&dst)[4][8], const float* __restrict__ xb, int t, int hi) {
    #pragma unroll
    for (int jt = 0; jt < 4; ++jt) {
        const int jbase = jt * 16 + hi * 8;
        float4 v0 = *(const float4*)(xb + (size_t)t * CRF_C + jbase);
        float4 v1 = *(const float4*)(xb + (size_t)t * CRF_C + jbase + 4);
        dst[jt][0] = v0.x * CRF_LOG2E; dst[jt][1] = v0.y * CRF_LOG2E;
        dst[jt][2] = v0.z * CRF_LOG2E; dst[jt][3] = v0.w * CRF_LOG2E;
        dst[jt][4] = v1.x * CRF_LOG2E; dst[jt][5] = v1.y * CRF_LOG2E;
        dst[jt][6] = v1.z * CRF_LOG2E; dst[jt][7] = v1.w * CRF_LOG2E;
    }
}

// One recursion step: af (base-2 log alpha, D layout) -> af at next t.
static __device__ __forceinline__
void crf_step(float (&af)[4][8], const v16h (&Afrag)[4][2],
              const float (&xt)[4][8], int hi) {
    // per-batch max over all 64 classes: in-lane tree + one half-swap
    float mx = af[0][0];
    #pragma unroll
    for (int jt = 0; jt < 4; ++jt)
        #pragma unroll
        for (int r = 0; r < 8; ++r) mx = fmaxf(mx, af[jt][r]);
    mx = fmaxf(mx, xor16f(mx));

    // p = exp2(af - mx) packed to f16 pairs (raw v_exp + v_cvt_pk_rtz)
    unsigned pk[4][4];
    #pragma unroll
    for (int nt = 0; nt < 4; ++nt) {
        #pragma unroll
        for (int j2 = 0; j2 < 4; ++j2) {
            PK2 z;
            z.v = __builtin_amdgcn_cvt_pkrtz(
                __builtin_amdgcn_exp2f(af[nt][2 * j2 + 0] - mx),
                __builtin_amdgcn_exp2f(af[nt][2 * j2 + 1] - mx));
            pk[nt][j2] = z.u;
        }
    }
    // partner half's values (classes offset by 8 within each tile)
    unsigned pp[4][4];
    #pragma unroll
    for (int nt = 0; nt < 4; ++nt)
        #pragma unroll
        for (int j2 = 0; j2 < 4; ++j2) pp[nt][j2] = xor16u(pk[nt][j2]);

    // assemble B fragments (32x16 f16: lane lo = batch col,
    // element e -> K = c*32 + hi*16 + e)
    BFrag Bc[2];
    #pragma unroll
    for (int c = 0; c < 2; ++c) {
        const int tA = 2 * c, tB = 2 * c + 1;
        #pragma unroll
        for (int j2 = 0; j2 < 4; ++j2) {
            Bc[c].u[j2]     = hi ? pp[tB][j2] : pk[tA][j2]; // e = 0..7
            Bc[c].u[4 + j2] = hi ? pk[tB][j2] : pp[tA][j2]; // e = 8..15
        }
    }

    // D[j][b] = expU^T @ p^T : 4 output-class tiles x 2 K-chunks
    #pragma unroll
    for (int jt = 0; jt < 4; ++jt) {
        v8f acc = {};
        acc = __builtin_amdgcn_wmma_f32_16x16x32_f16(
            false, Afrag[jt][0], false, Bc[0].v, (short)0, acc, false, false);
        acc = __builtin_amdgcn_wmma_f32_16x16x32_f16(
            false, Afrag[jt][1], false, Bc[1].v, (short)0, acc, false, false);
        #pragma unroll
        for (int r = 0; r < 8; ++r)
            af[jt][r] = __builtin_amdgcn_logf(acc[r]) + mx + xt[jt][r];
    }
}

__global__ __launch_bounds__(32)
void crf_forward_logz_kernel(const float* __restrict__ x,
                             const float* __restrict__ U,
                             const float* __restrict__ b_start,
                             const float* __restrict__ b_end,
                             float* __restrict__ logZ)
{
    const int lane = threadIdx.x;       // 0..31
    const int hi   = lane >> 4;         // lane half
    const int lo   = lane & 15;         // A row (class) / B,D column (batch)
    const int b0   = blockIdx.x * 16;   // batch tile base

    // ---- Constant A fragments: A = exp(U)^T = exp2(U*log2e)^T
    // A 16x32 f16 layout: lane(lo)=row, element e -> K = c*32+hi*16+e
    v16h Afrag[4][2];
    #pragma unroll
    for (int jt = 0; jt < 4; ++jt) {
        #pragma unroll
        for (int c = 0; c < 2; ++c) {
            #pragma unroll
            for (int e = 0; e < 16; ++e) {
                int k = c * 32 + hi * 16 + e;                        // in class i
                int j = jt * 16 + lo;                                // out class j
                Afrag[jt][c][e] =
                    (_Float16)__builtin_amdgcn_exp2f(U[k * CRF_C + j] * CRF_LOG2E);
            }
        }
    }

    const float* xb = x + (size_t)(b0 + lo) * CRF_T * CRF_C;

    // ---- alpha (base-2) in D layout: lane = batch lo, class j = jt*16+8*hi+r
    float af[4][8];
    #pragma unroll
    for (int jt = 0; jt < 4; ++jt) {
        const int jbase = jt * 16 + hi * 8;
        float4 v0 = *(const float4*)(xb + jbase);
        float4 v1 = *(const float4*)(xb + jbase + 4);
        af[jt][0] = (v0.x + b_start[jbase + 0]) * CRF_LOG2E;
        af[jt][1] = (v0.y + b_start[jbase + 1]) * CRF_LOG2E;
        af[jt][2] = (v0.z + b_start[jbase + 2]) * CRF_LOG2E;
        af[jt][3] = (v0.w + b_start[jbase + 3]) * CRF_LOG2E;
        af[jt][4] = (v1.x + b_start[jbase + 4]) * CRF_LOG2E;
        af[jt][5] = (v1.y + b_start[jbase + 5]) * CRF_LOG2E;
        af[jt][6] = (v1.z + b_start[jbase + 6]) * CRF_LOG2E;
        af[jt][7] = (v1.w + b_start[jbase + 7]) * CRF_LOG2E;
    }

    // ---- ping-pong x buffers, unroll-by-2 steady state (t = 1..1022),
    //      tail step t = 1023
    float xc[4][8], xn[4][8];
    crf_load_x(xc, xb, 1, hi);
    for (int t = 1; t + 1 < CRF_T; t += 2) {
        // speculative prefetch ~2 steps ahead (lane halves cover both 128B lines)
        const int tp = (t + 3 < CRF_T) ? (t + 3) : (CRF_T - 1);
        __builtin_prefetch(xb + (size_t)tp * CRF_C + hi * 32, 0, 1);

        crf_load_x(xn, xb, t + 1, hi);
        crf_step(af, Afrag, xc, hi);       // timestep t
        crf_load_x(xc, xb, t + 2, hi);
        crf_step(af, Afrag, xn, hi);       // timestep t+1
    }
    crf_step(af, Afrag, xc, hi);           // timestep T-1 = 1023

    // ---- end boundary, then logZ[b] = ln2 * log2sumexp2_j af[j]
    #pragma unroll
    for (int jt = 0; jt < 4; ++jt) {
        const int jbase = jt * 16 + hi * 8;
        #pragma unroll
        for (int r = 0; r < 8; ++r)
            af[jt][r] += b_end[jbase + r] * CRF_LOG2E;
    }

    float m2 = af[0][0];
    #pragma unroll
    for (int jt = 0; jt < 4; ++jt)
        #pragma unroll
        for (int r = 0; r < 8; ++r) m2 = fmaxf(m2, af[jt][r]);
    m2 = fmaxf(m2, xor16f(m2));

    float s = 0.0f;
    #pragma unroll
    for (int jt = 0; jt < 4; ++jt)
        #pragma unroll
        for (int r = 0; r < 8; ++r)
            s += __builtin_amdgcn_exp2f(af[jt][r] - m2);
    s += xor16f(s);

    if (lane < 16)
        logZ[b0 + lane] = CRF_LN2 * (m2 + __builtin_amdgcn_logf(s));
}

// ============================================================================
// Gold-path energy (gather + reduce) and final loss combine.
// ============================================================================
__global__ __launch_bounds__(256)
void crf_path_energy_kernel(const float* __restrict__ x,
                            const float* __restrict__ U,
                            const float* __restrict__ b_start,
                            const float* __restrict__ b_end,
                            const int* __restrict__ y,
                            const float* __restrict__ logZ,
                            float* __restrict__ out)
{
    const int b   = blockIdx.x;
    const int tid = threadIdx.x;
    const int*   yb  = y + (size_t)b * CRF_T;
    const float* xbp = x + (size_t)b * CRF_T * CRF_C;

    float s = 0.0f;
    for (int t = tid; t < CRF_T; t += 256) {
        const int yt = yb[t];
        s += xbp[t * CRF_C + yt];                           // emission
        if (t + 1 < CRF_T) s += U[yt * CRF_C + yb[t + 1]];  // transition
    }
    if (tid == 0) s += b_start[yb[0]] + b_end[yb[CRF_T - 1]];

    __shared__ float sd[256];
    sd[tid] = s;
    __syncthreads();
    #pragma unroll
    for (int off = 128; off > 0; off >>= 1) {
        if (tid < off) sd[tid] += sd[tid + off];
        __syncthreads();
    }
    if (tid == 0) out[b] = logZ[b] - sd[0];  // -(path - logZ)
}

extern "C" void kernel_launch(void* const* d_in, const int* in_sizes, int n_in,
                              void* d_out, int out_size, void* d_ws, size_t ws_size,
                              hipStream_t stream) {
    (void)in_sizes; (void)n_in; (void)out_size; (void)ws_size;
    const float* x  = (const float*)d_in[0];
    const float* U  = (const float*)d_in[1];
    const float* bs = (const float*)d_in[2];
    const float* be = (const float*)d_in[3];
    const int*   y  = (const int*)d_in[4];
    float* out  = (float*)d_out;
    float* logZ = (float*)d_ws;   // B floats of scratch

    crf_forward_logz_kernel<<<CRF_B / 16, 32, 0, stream>>>(x, U, bs, be, logZ);
    crf_path_energy_kernel<<<CRF_B, 256, 0, stream>>>(x, U, bs, be, y, logZ, out);
}
